// SupConTopKLoss_62809601737212
// MI455X (gfx1250) — matrix operand order repaired and checked by reading, not compiled, not run on hardware
//
#include <hip/hip_runtime.h>
#include <hip/hip_bf16.h>

typedef _Float16 v16h __attribute__((ext_vector_type(16)));
typedef _Float16 v8h  __attribute__((ext_vector_type(8)));
typedef float    v8f  __attribute__((ext_vector_type(8)));

#define BSZ        4096
#define DIM        128
#define TOPK       200
#define ROWS_PER_WG 16
#define WAVES       16
#define THREADS    (WAVES * 32)
#define NTILES     (BSZ / 16)

// ---------------------------------------------------------------------------
// Kernel A: split fp32 features into f16 hi/lo pair (error-compensated GEMM).
// ---------------------------------------------------------------------------
__global__ void supcon_split_kernel(const float* __restrict__ f,
                                    _Float16* __restrict__ hi,
                                    _Float16* __restrict__ lo, int n) {
    int i = blockIdx.x * blockDim.x + threadIdx.x;
    if (i < n) {
        float x = f[i];
        _Float16 h = (_Float16)x;
        hi[i] = h;
        lo[i] = (_Float16)(x - (float)h);
    }
}

// ---------------------------------------------------------------------------
// WMMA operand loaders (wave32, 16-bit operands).
// A 16x32 layout (ISA 7.12.2): lane m=lane&15 -> row M=m; lanes 0-15 carry
// K = {0..7, 16..23}, lanes 16-31 carry K = {8..15, 24..31}; two contiguous
// 16B runs per lane.
// ---------------------------------------------------------------------------
__device__ __forceinline__ v16h load_a_tile(const _Float16* __restrict__ base,
                                            int rowbase, int kb, int lane) {
    int m = lane & 15;
    int khalf = (lane >> 4) << 3;                 // 0 or 8
    const _Float16* p = base + (size_t)(rowbase + m) * DIM + kb + khalf;
    v8h x0 = *(const v8h*)p;                       // K = kb+khalf+0..7
    v8h x1 = *(const v8h*)(p + 16);                // K = kb+16+khalf+0..7
    return __builtin_shufflevector(x0, x1, 0, 1, 2, 3, 4, 5, 6, 7,
                                           8, 9, 10, 11, 12, 13, 14, 15);
}

// B 32x16 layout: lane n=lane&15 -> col N=n; lanes 0-15
// carry K = {0..7, 16..23}, lanes 16-31 carry K = {8..15, 24..31} (matches A).
// B[k][n] = F[col][k], so it loads straight from the row-major feature array.
__device__ __forceinline__ v16h load_b_tile(const _Float16* __restrict__ base,
                                            int colbase, int kb, int lane) {
    int n = lane & 15;
    const _Float16* p = base + (size_t)(colbase + n) * DIM + kb + ((lane >> 4) << 3);
    v8h x0 = *(const v8h*)p;
    v8h x1 = *(const v8h*)(p + 16);
    return __builtin_shufflevector(x0, x1, 0, 1, 2, 3, 4, 5, 6, 7,
                                           8, 9, 10, 11, 12, 13, 14, 15);
}

__device__ __forceinline__ float unkey(unsigned k) {
    unsigned u = (k >> 31) ? (k ^ 0x80000000u) : ~k;
    return __uint_as_float(u);
}

__device__ __forceinline__ float wave_red_add(float v) {
    for (int o = 16; o; o >>= 1) v += __shfl_xor(v, o, 32);
    return v;
}
__device__ __forceinline__ int wave_red_addi(int v) {
    for (int o = 16; o; o >>= 1) v += __shfl_xor(v, o, 32);
    return v;
}
__device__ __forceinline__ float wave_red_max(float v) {
    for (int o = 16; o; o >>= 1) v = fmaxf(v, __shfl_xor(v, o, 32));
    return v;
}

// ---------------------------------------------------------------------------
// Kernel B: one WG per 16 anchor rows.
//  Phase 1: S[16][4096] = F[r0:r0+16] . F^T via f16 WMMA (hi/lo split), -> LDS
//  Phase 2: one wave per row: row-max, positive sums, top-200 negative sum via
//           32-step binary search on order-preserving uint keys (in-place LDS).
// ---------------------------------------------------------------------------
__global__ void supcon_main_kernel(const _Float16* __restrict__ hi,
                                   const _Float16* __restrict__ lo,
                                   const int* __restrict__ labels,
                                   float* __restrict__ rowLoss,
                                   float* __restrict__ rowValid) {
    extern __shared__ char smem_raw[];
    float* S    = (float*)smem_raw;                                 // [16][4096]
    int*   labs = (int*)(smem_raw + (size_t)ROWS_PER_WG * BSZ * 4); // [4096]

    const int lane = threadIdx.x & 31;
    const int wave = threadIdx.x >> 5;
    const int r0 = blockIdx.x * ROWS_PER_WG;

    // Preload labels into LDS.
    for (int i = threadIdx.x; i < BSZ; i += THREADS) labs[i] = labels[i];

    // ---- Phase 1: WMMA Gram tile -------------------------------------------
    v16h a_hi[4], a_lo[4];
#pragma unroll
    for (int kk = 0; kk < 4; kk++) {
        a_hi[kk] = load_a_tile(hi, r0, kk * 32, lane);
        a_lo[kk] = load_a_tile(lo, r0, kk * 32, lane);
    }

    for (int t = wave; t < NTILES; t += WAVES) {
        int c0 = t * 16;
        v8f acc = {};
#pragma unroll
        for (int kk = 0; kk < 4; kk++) {
            v16h b_hi = load_b_tile(hi, c0, kk * 32, lane);
            v16h b_lo = load_b_tile(lo, c0, kk * 32, lane);
            acc = __builtin_amdgcn_wmma_f32_16x16x32_f16(
                false, a_hi[kk], false, b_hi, (short)0, acc, false, false);
            acc = __builtin_amdgcn_wmma_f32_16x16x32_f16(
                false, a_hi[kk], false, b_lo, (short)0, acc, false, false);
            acc = __builtin_amdgcn_wmma_f32_16x16x32_f16(
                false, a_lo[kk], false, b_hi, (short)0, acc, false, false);
        }
        // C/D layout: VGPR g -> M = g + 8*(lane>=16), N = lane&15.
        int n = lane & 15;
        int mofs = (lane >> 4) << 3;
#pragma unroll
        for (int g = 0; g < 8; g++) {
            S[(size_t)(g + mofs) * BSZ + c0 + n] = acc[g];
        }
    }
    __syncthreads();

    // ---- Phase 2: per-row reductions (wave w owns local row w) -------------
    const int row = r0 + wave;
    float* Srow = S + (size_t)wave * BSZ;
    unsigned* Krow = (unsigned*)Srow;
    const int lab_a = labs[row];

    // Row max (includes diagonal, as in the reference).
    float m = -3.0e38f;
    for (int j = lane; j < BSZ; j += 32) m = fmaxf(m, Srow[j]);
    m = wave_red_max(m);

    // Positive-class sums + in-place rewrite to sortable keys for negatives.
    float pcnt = 0.f, plog = 0.f, pexp = 0.f;
    for (int j = lane; j < BSZ; j += 32) {
        float s = Srow[j];
        int lj = labs[j];
        unsigned key = 0u;
        if (lj == lab_a) {
            if (j != row) {
                float lg = 10.f * (s - m);
                pcnt += 1.f;
                plog += lg;
                pexp += __expf(lg);
            }
        } else {
            unsigned u = __float_as_uint(s);
            key = (u >> 31) ? ~u : (u ^ 0x80000000u);
        }
        Krow[j] = key;     // each lane only ever re-reads its own indices
    }
    pcnt = wave_red_add(pcnt);
    plog = wave_red_add(plog);
    pexp = wave_red_add(pexp);

    // Binary search for the key of the 200th-largest negative value:
    // largest t with count(key >= t) >= TOPK.
    unsigned lo_k = 0u, hi_k = 0xFFFFFFFFu;
    while (lo_k < hi_k) {
        unsigned mid = (unsigned)((((unsigned long long)lo_k +
                                    (unsigned long long)hi_k + 1ull)) >> 1);
        int c = 0;
        for (int j = lane; j < BSZ; j += 32) c += (Krow[j] >= mid) ? 1 : 0;
        c = wave_red_addi(c);
        if (c >= TOPK) lo_k = mid; else hi_k = mid - 1u;
    }
    const unsigned Kstar = lo_k;

    // Sum exp over strictly-greater keys; add ties at the threshold so that
    // exactly TOPK entries contribute (matches scatter-of-indices semantics).
    float sumTop = 0.f;
    int cgt = 0;
    for (int j = lane; j < BSZ; j += 32) {
        unsigned k = Krow[j];
        if (k > Kstar) {
            sumTop += __expf(10.f * (unkey(k) - m));
            cgt++;
        }
    }
    sumTop = wave_red_add(sumTop);
    cgt = wave_red_addi(cgt);
    sumTop += (float)(TOPK - cgt) * __expf(10.f * (unkey(Kstar) - m));

    const float denom = pexp + sumTop;
    const float mean_lp = (plog - pcnt * __logf(denom)) / fmaxf(pcnt, 1.f);
    const float per_row = -2.0f * mean_lp;          // -(TEMP/BASE_TEMP)
    const bool valid = (lab_a > 0) && (pcnt > 0.f);

    if (lane == 0) {
        rowLoss[row]  = valid ? per_row : 0.f;
        rowValid[row] = valid ? 1.f : 0.f;
    }
}

// ---------------------------------------------------------------------------
// Kernel C: deterministic tree reduction of per-row results -> scalar loss.
// ---------------------------------------------------------------------------
__global__ void supcon_finalize_kernel(const float* __restrict__ rowLoss,
                                       const float* __restrict__ rowValid,
                                       float* __restrict__ out) {
    __shared__ float sl[256];
    __shared__ float sv[256];
    float a = 0.f, b = 0.f;
    for (int j = threadIdx.x; j < BSZ; j += 256) {
        a += rowLoss[j];
        b += rowValid[j];
    }
    sl[threadIdx.x] = a;
    sv[threadIdx.x] = b;
    __syncthreads();
    for (int s = 128; s; s >>= 1) {
        if (threadIdx.x < (unsigned)s) {
            sl[threadIdx.x] += sl[threadIdx.x + s];
            sv[threadIdx.x] += sv[threadIdx.x + s];
        }
        __syncthreads();
    }
    if (threadIdx.x == 0) out[0] = sl[0] / sv[0];
}

// ---------------------------------------------------------------------------
extern "C" void kernel_launch(void* const* d_in, const int* in_sizes, int n_in,
                              void* d_out, int out_size, void* d_ws, size_t ws_size,
                              hipStream_t stream) {
    const float* feats  = (const float*)d_in[0];   // [4096,1,128] f32
    const int*   labels = (const int*)d_in[1];     // [4096] i32
    float*       out    = (float*)d_out;           // scalar

    char* ws = (char*)d_ws;
    const size_t nelem = (size_t)BSZ * DIM;        // 524288
    _Float16* hi       = (_Float16*)ws;
    _Float16* lo       = (_Float16*)(ws + nelem * sizeof(_Float16));
    float*    rowLoss  = (float*)(ws + 2 * nelem * sizeof(_Float16));
    float*    rowValid = rowLoss + BSZ;

    // A: fp32 -> f16 hi/lo split
    supcon_split_kernel<<<(int)((nelem + 255) / 256), 256, 0, stream>>>(
        feats, hi, lo, (int)nelem);

    // B: fused Gram + top-k supcon per 16-row block (272 KB dynamic LDS)
    const size_t smem = (size_t)ROWS_PER_WG * BSZ * sizeof(float)
                      + (size_t)BSZ * sizeof(int);
    supcon_main_kernel<<<BSZ / ROWS_PER_WG, THREADS, smem, stream>>>(
        hi, lo, labels, rowLoss, rowValid);

    // C: deterministic final reduction
    supcon_finalize_kernel<<<1, 256, 0, stream>>>(rowLoss, rowValid, out);
}
